// GPTmodel_22454089024264
// MI455X (gfx1250) — matrix operand order repaired and checked by reading, not compiled, not run on hardware
//
#include <hip/hip_runtime.h>

typedef __attribute__((ext_vector_type(16))) __bf16 v16bf;
typedef __attribute__((ext_vector_type(4)))  __bf16 v4bf;
typedef __attribute__((ext_vector_type(8)))  float  v8f;

static constexpr int Bc   = 2;
static constexpr int Sc   = 1024;
static constexpr int Dc   = 1024;
static constexpr int DFFc = 4096;
static constexpr int Hc   = 16;
static constexpr int Vc   = 32000;
static constexpr int Lc   = 8;
static constexpr int Mc   = Bc * Sc;   // 2048 rows

static __device__ __forceinline__ __bf16 f2b(float f) { return (__bf16)f; }

// async 16-byte global -> LDS copy (ASYNCcnt-tracked, CDNA5)
static __device__ __forceinline__ void async_copy_b128(unsigned lds_off, const void* gaddr)
{
    asm volatile("global_load_async_to_lds_b128 %0, %1, off"
                 :: "v"(lds_off), "v"(gaddr) : "memory");
}
static __device__ __forceinline__ void wait_async0()
{
    asm volatile("s_wait_asynccnt 0x0" ::: "memory");
}

// ---------------------------------------------------------------------------
// Embedding * sqrt(D) + sinusoidal positional encoding (computed on device)
// ---------------------------------------------------------------------------
__global__ __launch_bounds__(256) void embed_kernel(const int* __restrict__ x,
                                                    const float* __restrict__ emb,
                                                    float* __restrict__ hout)
{
    int idx = blockIdx.x * 256 + threadIdx.x;      // 0 .. B*S*D-1
    int d   = idx & (Dc - 1);
    int bs  = idx >> 10;                           // D = 1024
    int s   = bs & (Sc - 1);
    int tok = x[bs];
    float div = __expf(-(float)(d & ~1) * (9.210340371976184f / (float)Dc));
    float ang = (float)s * div;
    float pe  = (d & 1) ? __cosf(ang) : __sinf(ang);
    hout[idx] = emb[(size_t)tok * Dc + d] * 32.0f + pe;   // sqrt(1024)=32
}

// ---------------------------------------------------------------------------
// One-time weight convert + transpose: W[K][N] fp32  ->  Wt[N][K] bf16
// ---------------------------------------------------------------------------
__global__ __launch_bounds__(256) void convT_kernel(const float* __restrict__ W,
                                                    __bf16* __restrict__ Wt,
                                                    int K, int N)
{
    __shared__ __bf16 t[32][33];
    const int k0 = blockIdx.x * 32;
    const int n0 = blockIdx.y * 32;
    const int c  = threadIdx.x & 31;
    const int r0 = threadIdx.x >> 5;               // 0..7
    for (int i = 0; i < 4; ++i) {
        int r = r0 + i * 8;
        t[r][c] = f2b(W[(size_t)(k0 + r) * N + n0 + c]);
    }
    __syncthreads();
    for (int i = 0; i < 4; ++i) {
        int r = r0 + i * 8;
        Wt[(size_t)(n0 + r) * K + k0 + c] = t[c][r];
    }
}

// ---------------------------------------------------------------------------
// LayerNorm: one block per row of D=1024; fp32 in, bf16 out (WMMA A operand)
// ---------------------------------------------------------------------------
__global__ __launch_bounds__(256) void ln_kernel(const float* __restrict__ X,
                                                 const float* __restrict__ g,
                                                 const float* __restrict__ bta,
                                                 __bf16* __restrict__ Y)
{
    __shared__ float red[256];
    const int row = blockIdx.x;
    const float* xr = X + (size_t)row * Dc;
    float local[4];
    float s = 0.f;
    for (int i = 0; i < 4; ++i) { local[i] = xr[threadIdx.x + 256 * i]; s += local[i]; }
    red[threadIdx.x] = s; __syncthreads();
    for (int st = 128; st > 0; st >>= 1) {
        if (threadIdx.x < st) red[threadIdx.x] += red[threadIdx.x + st];
        __syncthreads();
    }
    float mean = red[0] * (1.f / (float)Dc);
    __syncthreads();
    float vs = 0.f;
    for (int i = 0; i < 4; ++i) { float dd = local[i] - mean; vs += dd * dd; }
    red[threadIdx.x] = vs; __syncthreads();
    for (int st = 128; st > 0; st >>= 1) {
        if (threadIdx.x < st) red[threadIdx.x] += red[threadIdx.x + st];
        __syncthreads();
    }
    float inv = rsqrtf(red[0] * (1.f / (float)Dc) + 1e-5f);
    for (int i = 0; i < 4; ++i) {
        int c = threadIdx.x + 256 * i;
        Y[(size_t)row * Dc + c] = f2b((local[i] - mean) * inv * g[c] + bta[c]);
    }
}

// ---------------------------------------------------------------------------
// BF16 WMMA GEMM: C[M,N] = epilogue(A[M,K]bf16 @ Wt[N][K]bf16)
// 128x64 block tile, 128 threads = 4 wave32 (each wave 32 rows x 64 cols,
// 2 A-frags x 4 B-frags = 8 WMMA per K-step). Double-buffered LDS; BOTH
// operand tiles streamed via global_load_async_to_lds_b128 (ASYNCcnt).
// epilogue: (+bias) -> (relu) -> (+residual) -> store f32 (Cf) or bf16 (Cb)
// ---------------------------------------------------------------------------
__global__ __launch_bounds__(128) void gemm_kernel(const __bf16* __restrict__ A,
                                                   const __bf16* __restrict__ Wt,
                                                   const float* __restrict__ bias,
                                                   const float* R,
                                                   float* Cf, __bf16* Cb,
                                                   int M, int N, int Kd, int relu)
{
    __shared__ __bf16 As[2][128][32];  // [m][k]
    __shared__ __bf16 Bs[2][64][32];   // [n][k]
    const int n0   = blockIdx.x * 64;
    const int m0   = blockIdx.y * 128;
    const int tid  = threadIdx.x;
    const int lane = tid & 31;
    const int wv   = tid >> 5;
    const int hf   = lane >> 4;
    const int ln16 = lane & 15;

    v8f acc[2][4];
    for (int s2 = 0; s2 < 2; ++s2)
        for (int j = 0; j < 4; ++j)
            for (int e = 0; e < 8; ++e) acc[s2][j][e] = 0.f;

    const int nt = Kd >> 5;            // K tiles of 32

#define STAGE_A(T, BUF)                                                            \
    for (int it = 0; it < 4; ++it) {                                               \
        int cid = it * 128 + tid;       /* 0..511 chunks of 16B */                 \
        int ar = cid >> 2, co = (cid & 3) * 16;                                    \
        const char* g = (const char*)(A + (size_t)(m0 + ar) * Kd + (T) * 32) + co; \
        unsigned loff = (unsigned)(size_t)(&As[BUF][ar][0]) + co;                  \
        async_copy_b128(loff, g);                                                  \
    }
#define STAGE_B(T, BUF)                                                            \
    for (int c2 = 0; c2 < 2; ++c2) {                                               \
        int cid = c2 * 128 + tid;                                                  \
        int n = cid >> 2, co = (cid & 3) * 16;                                     \
        const char* g = (const char*)(Wt + (size_t)(n0 + n) * Kd + (T) * 32) + co; \
        unsigned loff = (unsigned)(size_t)(&Bs[BUF][n][0]) + co;                   \
        async_copy_b128(loff, g);                                                  \
    }

    STAGE_A(0, 0)
    STAGE_B(0, 0)
    wait_async0();
    __syncthreads();

    int buf = 0;
    for (int t = 0; t < nt; ++t) {
        if (t + 1 < nt) {
            STAGE_A(t + 1, buf ^ 1)
            STAGE_B(t + 1, buf ^ 1)
        }
        v16bf afrg[2];
        for (int s2 = 0; s2 < 2; ++s2) {
            int row = wv * 32 + s2 * 16 + ln16;
            int kb  = hf * 8;
            for (int p = 0; p < 8; ++p) {
                int base = (p < 4) ? 0 : 16;
                int tt = p & 3;
                int kl = base + kb + 2 * tt;
                afrg[s2][2 * p]     = As[buf][row][kl];
                afrg[s2][2 * p + 1] = As[buf][row][kl + 1];
            }
        }
        for (int j = 0; j < 4; ++j) {
            v16bf bw;
            int n  = j * 16 + ln16;
            int kb = hf * 8;
            for (int p = 0; p < 8; ++p) {
                int kl = ((p < 4) ? 0 : 16) + kb + 2 * (p & 3);
                bw[2 * p]     = Bs[buf][n][kl];
                bw[2 * p + 1] = Bs[buf][n][kl + 1];
            }
            acc[0][j] = __builtin_amdgcn_wmma_f32_16x16x32_bf16(
                false, afrg[0], false, bw, (short)0, acc[0][j], false, false);
            acc[1][j] = __builtin_amdgcn_wmma_f32_16x16x32_bf16(
                false, afrg[1], false, bw, (short)0, acc[1][j], false, false);
        }
        wait_async0();
        __syncthreads();
        buf ^= 1;
    }
#undef STAGE_A
#undef STAGE_B

    for (int s2 = 0; s2 < 2; ++s2) {
        for (int j = 0; j < 4; ++j) {
            int ng = n0 + j * 16 + ln16;
            float bv = bias ? bias[ng] : 0.f;
            for (int r = 0; r < 8; ++r) {
                int mg = m0 + wv * 32 + s2 * 16 + r + hf * 8;
                float val = acc[s2][j][r] + bv;
                if (relu) val = fmaxf(val, 0.f);
                if (R) val += R[(size_t)mg * N + ng];
                if (Cb) Cb[(size_t)mg * N + ng] = f2b(val);
                else    Cf[(size_t)mg * N + ng] = val;
            }
        }
    }
}

// ---------------------------------------------------------------------------
// Flash-style causal attention with WMMA (bf16 in, bf16 out, fp32 softmax)
// grid: (S/64, B*H); block: 128 threads (4 waves, 16 query rows each)
// Q/K tiles async-copied; V manually transposed into LDS.
// ---------------------------------------------------------------------------
__global__ __launch_bounds__(128) void attn_kernel(const __bf16* __restrict__ Q,
                                                   const __bf16* __restrict__ K,
                                                   const __bf16* __restrict__ V,
                                                   __bf16* __restrict__ O)
{
    __shared__ __bf16 Qs[64][64];        // [q_local][dh]
    __shared__ __bf16 Ks[64][64];        // [k_local][dh]
    __shared__ __bf16 Vs[64][64];        // [dh][k_local]  (transposed)
    __shared__ __bf16 Ps[4][16][64];     // per-wave P staging [wave][m][k_local]

    const int qt   = blockIdx.x;
    const int bh   = blockIdx.y;
    const int b    = bh / Hc;
    const int hh   = bh % Hc;
    const int tid  = threadIdx.x;
    const int lane = tid & 31;
    const int wv   = tid >> 5;
    const int hf   = lane >> 4;
    const int ln16 = lane & 15;
    const float scale = 0.125f;          // 1/sqrt(64)

    // Q tile: pure async byte copy (row = 64 bf16 = 128B = 8 chunks)
    for (int it = 0; it < 4; ++it) {
        int cid = it * 128 + tid;        // 0..511
        int row = cid >> 3, co = (cid & 7) * 16;
        const char* g = (const char*)(Q + ((size_t)b * Sc + qt * 64 + row) * Dc + hh * 64) + co;
        unsigned loff = (unsigned)(size_t)(&Qs[row][0]) + co;
        async_copy_b128(loff, g);
    }

    float mstate[8], lstate[8];
    v8f oacc[4];
    for (int r = 0; r < 8; ++r) { mstate[r] = -3.0e38f; lstate[r] = 0.f; }
    for (int j = 0; j < 4; ++j)
        for (int e = 0; e < 8; ++e) oacc[j][e] = 0.f;

    for (int kt = 0; kt <= qt; ++kt) {
        __syncthreads();
        // K tile: async copy; V tile: manual transpose
        for (int it = 0; it < 4; ++it) {
            int cid = it * 128 + tid;
            int row = cid >> 3, co = (cid & 7) * 16;
            const char* g = (const char*)(K + ((size_t)b * Sc + kt * 64 + row) * Dc + hh * 64) + co;
            unsigned loff = (unsigned)(size_t)(&Ks[row][0]) + co;
            async_copy_b128(loff, g);
        }
        for (int it = 0; it < 8; ++it) {
            int cid = it * 128 + tid;
            int row = cid >> 4, colc = (cid & 15) * 4;
            v4bf fv = *(const v4bf*)(V + ((size_t)b * Sc + kt * 64 + row) * Dc + hh * 64 + colc);
            Vs[colc + 0][row] = fv[0];
            Vs[colc + 1][row] = fv[1];
            Vs[colc + 2][row] = fv[2];
            Vs[colc + 3][row] = fv[3];
        }
        wait_async0();
        __syncthreads();

        // --- S = Q @ K^T (16x64 per wave) ---
        v8f sacc[4];
        for (int j = 0; j < 4; ++j)
            for (int e = 0; e < 8; ++e) sacc[j][e] = 0.f;
        for (int c = 0; c < 2; ++c) {
            v16bf aq;
            {
                int row = wv * 16 + ln16;
                int kb = hf * 8;
                for (int p = 0; p < 8; ++p) {
                    int base = (p < 4) ? 0 : 16;
                    int t = p & 3;
                    int kl = c * 32 + base + kb + 2 * t;
                    aq[2 * p]     = Qs[row][kl];
                    aq[2 * p + 1] = Qs[row][kl + 1];
                }
            }
            for (int j = 0; j < 4; ++j) {
                v16bf bk;
                int n  = j * 16 + ln16;
                int kb = hf * 8;
                for (int p = 0; p < 8; ++p) {
                    int kl = c * 32 + ((p < 4) ? 0 : 16) + kb + 2 * (p & 3);
                    bk[2 * p]     = Ks[n][kl];   // K^T[dh][key] == Ks[key][dh]
                    bk[2 * p + 1] = Ks[n][kl + 1];
                }
                sacc[j] = __builtin_amdgcn_wmma_f32_16x16x32_bf16(
                    false, aq, false, bk, (short)0, sacc[j], false, false);
            }
        }

        // --- causal mask + scale, online softmax (rows in 16-lane halves) ---
        float pv[4][8];
        for (int j = 0; j < 4; ++j) {
            int keyg = kt * 64 + j * 16 + ln16;
            for (int r = 0; r < 8; ++r) {
                int qg = qt * 64 + wv * 16 + r + hf * 8;
                float sv = sacc[j][r] * scale;
                if (keyg > qg) sv = -1.0e9f;
                pv[j][r] = sv;
            }
        }
        for (int r = 0; r < 8; ++r) {
            float rm = fmaxf(fmaxf(pv[0][r], pv[1][r]), fmaxf(pv[2][r], pv[3][r]));
            for (int m = 1; m < 16; m <<= 1) rm = fmaxf(rm, __shfl_xor(rm, m, 32));
            float nm    = fmaxf(mstate[r], rm);
            float alpha = __expf(mstate[r] - nm);
            mstate[r]   = nm;
            float rs = 0.f;
            for (int j = 0; j < 4; ++j) {
                float p = __expf(pv[j][r] - nm);
                pv[j][r] = p; rs += p;
            }
            for (int m = 1; m < 16; m <<= 1) rs += __shfl_xor(rs, m, 32);
            lstate[r] = lstate[r] * alpha + rs;
            for (int j = 0; j < 4; ++j) oacc[j][r] *= alpha;
        }

        // stage P to per-wave LDS (C-layout -> A-layout reshape)
        for (int j = 0; j < 4; ++j)
            for (int r = 0; r < 8; ++r)
                Ps[wv][r + hf * 8][j * 16 + ln16] = f2b(pv[j][r]);

        // --- O += P @ V ---
        for (int c = 0; c < 2; ++c) {
            v16bf ap;
            {
                int row = ln16;
                int kb = hf * 8;
                for (int p = 0; p < 8; ++p) {
                    int base = (p < 4) ? 0 : 16;
                    int t = p & 3;
                    int kl = c * 32 + base + kb + 2 * t;
                    ap[2 * p]     = Ps[wv][row][kl];
                    ap[2 * p + 1] = Ps[wv][row][kl + 1];
                }
            }
            for (int j = 0; j < 4; ++j) {
                v16bf bv;
                int dh = j * 16 + ln16;
                int kb = hf * 8;
                for (int p = 0; p < 8; ++p) {
                    int kl = c * 32 + ((p < 4) ? 0 : 16) + kb + 2 * (p & 3);
                    bv[2 * p]     = Vs[dh][kl];
                    bv[2 * p + 1] = Vs[dh][kl + 1];
                }
                oacc[j] = __builtin_amdgcn_wmma_f32_16x16x32_bf16(
                    false, ap, false, bv, (short)0, oacc[j], false, false);
            }
        }
    }

    // normalize + write (bf16)
    for (int j = 0; j < 4; ++j) {
        int dh = j * 16 + ln16;
        for (int r = 0; r < 8; ++r) {
            int sg = qt * 64 + wv * 16 + r + hf * 8;
            O[((size_t)b * Sc + sg) * Dc + hh * 64 + dh] = f2b(oacc[j][r] / lstate[r]);
        }
    }
}

// ---------------------------------------------------------------------------
extern "C" void kernel_launch(void* const* d_in, const int* in_sizes, int n_in,
                              void* d_out, int out_size, void* d_ws, size_t ws_size,
                              hipStream_t stream)
{
    (void)in_sizes; (void)n_in; (void)out_size; (void)ws_size;
    const int*   x    = (const int*)d_in[0];
    // d_in[1] = causal mask (recomputed on device)
    const float* emb  = (const float*)d_in[2];
    const float* Wq   = (const float*)d_in[3];
    const float* Wk   = (const float*)d_in[4];
    const float* Wv   = (const float*)d_in[5];
    const float* Wo   = (const float*)d_in[6];
    const float* ln1g = (const float*)d_in[7];
    const float* ln1b = (const float*)d_in[8];
    const float* ln2g = (const float*)d_in[9];
    const float* ln2b = (const float*)d_in[10];
    const float* W1   = (const float*)d_in[11];
    const float* b1   = (const float*)d_in[12];
    const float* W2   = (const float*)d_in[13];
    const float* b2   = (const float*)d_in[14];
    const float* lnfg = (const float*)d_in[15];
    const float* lnfb = (const float*)d_in[16];
    const float* pW   = (const float*)d_in[17];
    const float* pb   = (const float*)d_in[18];
    float* out = (float*)d_out;

    // fp32 residual stream, then bf16 activation buffers, then bf16 weights
    float*  h    = (float*)d_ws;
    __bf16* lnb  = (__bf16*)(h + (size_t)Mc * Dc);
    __bf16* q    = lnb + (size_t)Mc * Dc;
    __bf16* k    = q   + (size_t)Mc * Dc;
    __bf16* v    = k   + (size_t)Mc * Dc;
    __bf16* att  = v   + (size_t)Mc * Dc;
    __bf16* ff   = att + (size_t)Mc * Dc;
    __bf16* pool = ff  + (size_t)Mc * DFFc;

    size_t poff = 0;
    __bf16 *WqT[Lc], *WkT[Lc], *WvT[Lc], *WoT[Lc], *W1T[Lc], *W2T[Lc];
    for (int i = 0; i < Lc; ++i) {
        WqT[i] = pool + poff; poff += (size_t)Dc * Dc;
        WkT[i] = pool + poff; poff += (size_t)Dc * Dc;
        WvT[i] = pool + poff; poff += (size_t)Dc * Dc;
        WoT[i] = pool + poff; poff += (size_t)Dc * Dc;
        W1T[i] = pool + poff; poff += (size_t)Dc * DFFc;
        W2T[i] = pool + poff; poff += (size_t)DFFc * Dc;
    }
    __bf16* pWT = pool + poff;

    const dim3 blk128(128), blk256(256);
    const dim3 gD(Dc / 64, Mc / 128);     // N=1024
    const dim3 gF(DFFc / 64, Mc / 128);   // N=4096
    const dim3 gV(Vc / 64, Mc / 128);     // N=32000
    const dim3 gA(Sc / 64, Bc * Hc);

    // one-time weight convert+transpose to bf16 [N][K]
    for (int i = 0; i < Lc; ++i) {
        const size_t wd = (size_t)i * Dc * Dc;
        const size_t wf = (size_t)i * Dc * DFFc;
        convT_kernel<<<dim3(Dc / 32, Dc / 32),   blk256, 0, stream>>>(Wq + wd, WqT[i], Dc,   Dc);
        convT_kernel<<<dim3(Dc / 32, Dc / 32),   blk256, 0, stream>>>(Wk + wd, WkT[i], Dc,   Dc);
        convT_kernel<<<dim3(Dc / 32, Dc / 32),   blk256, 0, stream>>>(Wv + wd, WvT[i], Dc,   Dc);
        convT_kernel<<<dim3(Dc / 32, Dc / 32),   blk256, 0, stream>>>(Wo + wd, WoT[i], Dc,   Dc);
        convT_kernel<<<dim3(Dc / 32, DFFc / 32), blk256, 0, stream>>>(W1 + wf, W1T[i], Dc,   DFFc);
        convT_kernel<<<dim3(DFFc / 32, Dc / 32), blk256, 0, stream>>>(W2 + wf, W2T[i], DFFc, Dc);
    }
    convT_kernel<<<dim3(Dc / 32, Vc / 32), blk256, 0, stream>>>(pW, pWT, Dc, Vc);

    embed_kernel<<<(Mc * Dc) / 256, blk256, 0, stream>>>(x, emb, h);

    for (int i = 0; i < Lc; ++i) {
        ln_kernel<<<Mc, blk256, 0, stream>>>(h, ln1g + i * Dc, ln1b + i * Dc, lnb);
        gemm_kernel<<<gD, blk128, 0, stream>>>(lnb, WqT[i], nullptr, nullptr, nullptr, q, Mc, Dc, Dc, 0);
        gemm_kernel<<<gD, blk128, 0, stream>>>(lnb, WkT[i], nullptr, nullptr, nullptr, k, Mc, Dc, Dc, 0);
        gemm_kernel<<<gD, blk128, 0, stream>>>(lnb, WvT[i], nullptr, nullptr, nullptr, v, Mc, Dc, Dc, 0);
        attn_kernel<<<gA, blk128, 0, stream>>>(q, k, v, att);
        gemm_kernel<<<gD, blk128, 0, stream>>>(att, WoT[i], nullptr, h, h, nullptr, Mc, Dc, Dc, 0);
        ln_kernel<<<Mc, blk256, 0, stream>>>(h, ln2g + i * Dc, ln2b + i * Dc, lnb);
        gemm_kernel<<<gF, blk128, 0, stream>>>(lnb, W1T[i], b1 + i * DFFc, nullptr, nullptr, ff, Mc, DFFc, Dc, 1);
        gemm_kernel<<<gD, blk128, 0, stream>>>(ff, W2T[i], b2 + i * Dc, h, h, nullptr, Mc, Dc, DFFc, 0);
    }
    ln_kernel<<<Mc, blk256, 0, stream>>>(h, lnfg, lnfb, lnb);
    gemm_kernel<<<gV, blk128, 0, stream>>>(lnb, pWT, pb, nullptr, out, nullptr, Mc, Vc, Dc, 0);
}
